// LLaMAEncoder_53970559041866
// MI455X (gfx1250) — matrix operand order repaired and checked by reading, not compiled, not run on hardware
//
#include <hip/hip_runtime.h>
#include <hip/hip_bf16.h>

// ============================================================================
// QLoRA-LLaMA forward for MI455X (gfx1250).
// Compute-bound (≈456 GFLOP vs <0.3 GB traffic): all large GEMMs go through
// v_wmma_f32_16x16x32_bf16 (wave32, 16x16x32). NF4 weights are dequantized to
// bf16 once per qlinear into scratch. Attention is flash-style: one wave per
// 16-row Q tile, 32-key streaming tiles, WMMA for both Q·K^T and P·V, P tile
// re-laid-out via LDS (C-frag -> A-frag).
//
// d_in order (top-level dict order, nested dicts flattened alphabetically,
// lists in order; 'A' < 'B' < 'absmax' since uppercase sorts first):
//   0 points  1 attn_mask  2 params.final_ln
//   3+23*li .. layer li params: down{A,B,absmax} gate{A,B,absmax} k{A,B,absmax}
//              ln1 ln2 o{A,B,absmax} q{A,B,absmax} up{A,B,absmax} v{A,B,absmax}
//   49..55 tok: B_fourier b1 b2 ln_b ln_g w1 w2
//   56+7*li .. layer li codes: down gate k o q up v
// ============================================================================

typedef __attribute__((ext_vector_type(16))) __bf16 v16bf;
typedef __attribute__((ext_vector_type(8)))  float  v8f;

union FragAB {
    v16bf v;
    uint4 q[2];
    unsigned int u[8];
};

__device__ __constant__ float NF4_TAB[16] = {
    -1.0f, -0.6961928009986877f, -0.5250730514526367f, -0.39491748809814453f,
    -0.28444138169288635f, -0.18477343022823334f, -0.09105003625154495f, 0.0f,
    0.07958029955625534f, 0.16093020141124725f, 0.24611230194568634f,
    0.33791524171829224f, 0.44070982933044434f, 0.5626170039176941f,
    0.7229568362236023f, 1.0f};

__device__ inline unsigned short f2bf(float f) {
    unsigned int u = __float_as_uint(f);
    u += 0x7FFFu + ((u >> 16) & 1u);   // round-to-nearest-even
    return (unsigned short)(u >> 16);
}

// A-fragment (16x32 bf16, row-major src, leading dim ld in elements).
// lanes 0-15: row=lane, K 0..15; lanes 16-31: same rows, K 16..31.
__device__ inline void load_frag_a(FragAB& f, const unsigned short* base, int ld, int lane) {
    int m = lane & 15, sel = (lane >> 4) & 1;
    const unsigned short* p = base + (size_t)m * ld + sel * 16;
    f.q[0] = *(const uint4*)(p);
    f.q[1] = *(const uint4*)(p + 8);
}

// B-fragment (32x16 bf16) where column n is CONTIGUOUS in memory (row n of W,
// since B = W^T). lanes 0-15: K=0..15, lanes 16-31: K=16..31.
__device__ inline void load_frag_b_cols(FragAB& f, const unsigned short* base, int ld, int lane) {
    int n = lane & 15, kh = (lane >> 4) & 1;
    const unsigned short* p = base + (size_t)n * ld + kh * 16;
    f.q[0] = *(const uint4*)(p);
    f.q[1] = *(const uint4*)(p + 8);
}

// B-fragment (32x16 bf16) where element (k,n) = base[k*ld + n] (strided cols).
__device__ inline void load_frag_b_strided(FragAB& f, const unsigned short* base, int ld, int lane) {
    int n = lane & 15, kh = (lane >> 4) & 1;
    const unsigned short* p = base + (size_t)(kh * 16) * ld + n;
#pragma unroll
    for (int t = 0; t < 8; t++) {
        unsigned int lo = p[(size_t)(2 * t) * ld];
        unsigned int hi = p[(size_t)(2 * t + 1) * ld];
        f.u[t] = lo | (hi << 16);
    }
}

__device__ inline v8f wmma_bf16(const FragAB& a, const FragAB& b, v8f c) {
    return __builtin_amdgcn_wmma_f32_16x16x32_bf16(false, a.v, false, b.v,
                                                   (short)0, c, false, false);
}

// ---------------------------------------------------------------------------
// NF4 dequant: codes[o,i] (int 0..15) * absmax[o, i/64] -> bf16 W[o,i]
// ---------------------------------------------------------------------------
__global__ void __launch_bounds__(256) dequant_nf4_kernel(
    const int* __restrict__ codes, const float* __restrict__ absmax,
    unsigned short* __restrict__ W, int outd, int ind, size_t total) {
    size_t idx = (size_t)blockIdx.x * 256 + threadIdx.x;
    if (idx >= total) return;
    int row = (int)(idx / ind);
    int col = (int)(idx % ind);
    float w = NF4_TAB[codes[idx] & 15] * absmax[(size_t)row * (ind >> 6) + (col >> 6)];
    W[idx] = f2bf(w);
}

// ---------------------------------------------------------------------------
// C[M,N] = A[M,K](bf16 rowmajor) x W[N,K]^T (bf16 rowmajor) ; f32 out.
// One wave per block; each wave owns a 32x64 C tile (2x4 wmma accumulators).
// ---------------------------------------------------------------------------
__global__ void __launch_bounds__(32) gemm_bf16_nt_kernel(
    const unsigned short* __restrict__ A, const unsigned short* __restrict__ W,
    float* __restrict__ C, int M, int N, int K) {
    int lane = threadIdx.x;
    int n0 = blockIdx.x * 64;
    int m0 = blockIdx.y * 32;
    v8f acc[2][4];
#pragma unroll
    for (int i = 0; i < 2; i++)
#pragma unroll
        for (int j = 0; j < 4; j++)
            acc[i][j] = v8f{0.f, 0.f, 0.f, 0.f, 0.f, 0.f, 0.f, 0.f};

    for (int k0 = 0; k0 < K; k0 += 32) {
        __builtin_prefetch(A + (size_t)m0 * K + k0 + 64, 0, 1);
        FragAB fa[2], fb[4];
        load_frag_a(fa[0], A + (size_t)m0 * K + k0, K, lane);
        load_frag_a(fa[1], A + (size_t)(m0 + 16) * K + k0, K, lane);
#pragma unroll
        for (int j = 0; j < 4; j++)
            load_frag_b_cols(fb[j], W + (size_t)(n0 + j * 16) * K + k0, K, lane);
#pragma unroll
        for (int i = 0; i < 2; i++)
#pragma unroll
            for (int j = 0; j < 4; j++)
                acc[i][j] = wmma_bf16(fa[i], fb[j], acc[i][j]);
    }
    int n = lane & 15, sel = (lane >> 4) & 1;
#pragma unroll
    for (int i = 0; i < 2; i++)
#pragma unroll
        for (int j = 0; j < 4; j++)
#pragma unroll
            for (int r = 0; r < 8; r++)
                C[(size_t)(m0 + i * 16 + r + sel * 8) * N + n0 + j * 16 + n] = acc[i][j][r];
}

// ---------------------------------------------------------------------------
// LoRA: t16[m,r] = sum_k x[m,k]*A[r,k]   (r=16)
// ---------------------------------------------------------------------------
__global__ void __launch_bounds__(256) lora1_kernel(
    const float* __restrict__ x, const float* __restrict__ A,
    float* __restrict__ t16, int K) {
    int m = blockIdx.x, tid = threadIdx.x;
    int r = tid >> 4, sub = tid & 15;
    const float* xr = x + (size_t)m * K;
    const float* ar = A + (size_t)r * K;
    float s = 0.f;
    for (int k = sub; k < K; k += 16) s += xr[k] * ar[k];
#pragma unroll
    for (int off = 1; off < 16; off <<= 1) s += __shfl_xor(s, off, 16);
    if (sub == 0) t16[m * 16 + r] = s;
}

// y[m,n] += 2.0 * sum_r t16[m,r]*B[n,r]
__global__ void __launch_bounds__(256) lora2_kernel(
    const float* __restrict__ t16, const float* __restrict__ Bm,
    float* __restrict__ y, int N, size_t total) {
    size_t idx = (size_t)blockIdx.x * 256 + threadIdx.x;
    if (idx >= total) return;
    int m = (int)(idx / N), n = (int)(idx % N);
    const float* tr = t16 + (size_t)m * 16;
    const float* br = Bm + (size_t)n * 16;
    float s = 0.f;
#pragma unroll
    for (int r = 0; r < 16; r++) s += tr[r] * br[r];
    y[idx] += 2.0f * s;
}

// ---------------------------------------------------------------------------
// RMSNorm (block per row)
// ---------------------------------------------------------------------------
__global__ void __launch_bounds__(256) rmsnorm_kernel(
    const float* __restrict__ x, const float* __restrict__ g,
    float* __restrict__ y, int D) {
    __shared__ float red[256];
    int row = blockIdx.x, t = threadIdx.x;
    const float* xr = x + (size_t)row * D;
    float s = 0.f;
    for (int i = t; i < D; i += 256) { float v = xr[i]; s += v * v; }
    red[t] = s; __syncthreads();
    for (int o = 128; o > 0; o >>= 1) { if (t < o) red[t] += red[t + o]; __syncthreads(); }
    float inv = rsqrtf(red[0] / (float)D + 1e-5f);
    float* yr = y + (size_t)row * D;
    for (int i = t; i < D; i += 256) yr[i] = xr[i] * inv * g[i];
}

// f32 -> bf16 convert
__global__ void __launch_bounds__(256) conv_bf16_kernel(
    const float* __restrict__ x, unsigned short* __restrict__ y, size_t total) {
    size_t idx = (size_t)blockIdx.x * 256 + threadIdx.x;
    if (idx < total) y[idx] = f2bf(x[idx]);
}

__global__ void __launch_bounds__(256) add_kernel(
    float* __restrict__ x, const float* __restrict__ y, size_t total) {
    size_t idx = (size_t)blockIdx.x * 256 + threadIdx.x;
    if (idx < total) x[idx] += y[idx];
}

// h = silu(g) * u  (into g)
__global__ void __launch_bounds__(256) silu_mul_kernel(
    float* __restrict__ g, const float* __restrict__ u, size_t total) {
    size_t idx = (size_t)blockIdx.x * 256 + threadIdx.x;
    if (idx >= total) return;
    float gv = g[idx];
    g[idx] = (gv / (1.f + __expf(-gv))) * u[idx];
}

// ---------------------------------------------------------------------------
// RoPE over q,k (in place, pair-safe): one thread per (token, head, i<64)
// ---------------------------------------------------------------------------
__global__ void __launch_bounds__(256) rope_kernel(
    float* __restrict__ q, float* __restrict__ k, int N, int D, size_t totalPairs) {
    size_t idx = (size_t)blockIdx.x * 256 + threadIdx.x;
    if (idx >= totalPairs) return;
    int m = (int)(idx / (D / 2));
    int t = (int)(idx % (D / 2));
    int h = t >> 6, i = t & 63;
    int pos = m % N;
    float ang = (float)pos * __expf(-(float)i * (9.210340371976184f / 64.f));
    float sn, cs;
    __sincosf(ang, &sn, &cs);
    size_t c1 = (size_t)m * D + h * 128 + i, c2 = c1 + 64;
    float q1 = q[c1], q2 = q[c2], k1 = k[c1], k2 = k[c2];
    q[c1] = q1 * cs - q2 * sn; q[c2] = q2 * cs + q1 * sn;
    k[c1] = k1 * cs - k2 * sn; k[c2] = k2 * cs + k1 * sn;
}

// ---------------------------------------------------------------------------
// Flash attention: one wave per (b, h, 16-row q tile). Streams 32-key tiles,
// WMMA for Q·K^T (bf16) and P·V (bf16, P via LDS relayout), running softmax.
// ---------------------------------------------------------------------------
__global__ void __launch_bounds__(32) flash_attn_kernel(
    const unsigned short* __restrict__ Q, const unsigned short* __restrict__ K,
    const unsigned short* __restrict__ V, const int* __restrict__ mask,
    float* __restrict__ O, int N, int Dm) {
    __shared__ unsigned short sP[16 * 32];
    int lane = threadIdx.x;
    int q0 = blockIdx.x * 16;
    int h  = blockIdx.y;
    int b  = blockIdx.z;
    size_t base = (size_t)b * N * Dm + (size_t)h * 128;
    const unsigned short* Qh = Q + base;
    const unsigned short* Kh = K + base;
    const unsigned short* Vh = V + base;

    FragAB fq[4];
#pragma unroll
    for (int c = 0; c < 4; c++)
        load_frag_a(fq[c], Qh + (size_t)q0 * Dm + c * 32, Dm, lane);

    float m_run[8], l_run[8];
    v8f accO[8];
#pragma unroll
    for (int r = 0; r < 8; r++) { m_run[r] = -1e30f; l_run[r] = 0.f; }
#pragma unroll
    for (int j = 0; j < 8; j++) accO[j] = v8f{0.f, 0.f, 0.f, 0.f, 0.f, 0.f, 0.f, 0.f};

    int ncol = lane & 15, sel = (lane >> 4) & 1;
    const float scale = 0.08838834764831845f;  // 1/sqrt(128)
    int kend = q0 + 16;

    for (int k0 = 0; k0 < kend; k0 += 32) {
        v8f S0 = v8f{0.f, 0.f, 0.f, 0.f, 0.f, 0.f, 0.f, 0.f};
        v8f S1 = v8f{0.f, 0.f, 0.f, 0.f, 0.f, 0.f, 0.f, 0.f};
        FragAB fk;
#pragma unroll
        for (int c = 0; c < 4; c++) {
            load_frag_b_cols(fk, Kh + (size_t)k0 * Dm + c * 32, Dm, lane);
            S0 = wmma_bf16(fq[c], fk, S0);
        }
        if (k0 + 16 < kend) {
#pragma unroll
            for (int c = 0; c < 4; c++) {
                load_frag_b_cols(fk, Kh + (size_t)(k0 + 16) * Dm + c * 32, Dm, lane);
                S1 = wmma_bf16(fq[c], fk, S1);
            }
        }
        int j0 = k0 + ncol, j1 = k0 + 16 + ncol;
        bool am0 = mask[(size_t)b * N + j0] > 0;
        bool am1 = (j1 < N) && (mask[(size_t)b * N + j1] > 0);
#pragma unroll
        for (int r = 0; r < 8; r++) {
            int row = q0 + r + sel * 8;
            float s0 = (j0 <= row && am0) ? S0[r] * scale : -1e30f;
            float s1 = (j1 <= row && am1) ? S1[r] * scale : -1e30f;
            float mx = fmaxf(s0, s1);
#pragma unroll
            for (int off = 1; off < 16; off <<= 1) mx = fmaxf(mx, __shfl_xor(mx, off, 16));
            float mn = fmaxf(m_run[r], mx);
            float alpha = __expf(m_run[r] - mn);
            float p0 = __expf(s0 - mn);
            float p1 = __expf(s1 - mn);
            float rs = p0 + p1;
#pragma unroll
            for (int off = 1; off < 16; off <<= 1) rs += __shfl_xor(rs, off, 16);
            l_run[r] = l_run[r] * alpha + rs;
            m_run[r] = mn;
#pragma unroll
            for (int j = 0; j < 8; j++) accO[j][r] *= alpha;
            int rl = r + sel * 8;
            sP[rl * 32 + ncol]      = f2bf(p0);
            sP[rl * 32 + 16 + ncol] = f2bf(p1);
        }
        __syncthreads();
        FragAB fp;
        load_frag_a(fp, sP, 32, lane);
#pragma unroll
        for (int j = 0; j < 8; j++) {
            FragAB fv;
            load_frag_b_strided(fv, Vh + (size_t)k0 * Dm + j * 16, Dm, lane);
            accO[j] = wmma_bf16(fp, fv, accO[j]);
        }
        __syncthreads();
    }
    float* Oh = O + base;
#pragma unroll
    for (int j = 0; j < 8; j++)
#pragma unroll
        for (int r = 0; r < 8; r++)
            Oh[(size_t)(q0 + r + sel * 8) * Dm + j * 16 + ncol] = accO[j][r] / l_run[r];
}

// ---------------------------------------------------------------------------
// Point tokenizer: per-token MLP + layernorm. Block (256) per token.
// ---------------------------------------------------------------------------
__global__ void __launch_bounds__(256) tokenizer_kernel(
    const float* __restrict__ pts, const float* __restrict__ Bf,
    const float* __restrict__ w1, const float* __restrict__ b1,
    const float* __restrict__ w2, const float* __restrict__ b2,
    const float* __restrict__ lng, const float* __restrict__ lnb,
    float* __restrict__ xout) {
    __shared__ float sx[67];
    __shared__ float sh1[256];
    __shared__ float red[256];
    int tok = blockIdx.x, t = threadIdx.x;
    const float* p = pts + (size_t)tok * 3;
    if (t < 3) sx[t] = p[t];
    if (t < 32) {
        float pr = p[0] * Bf[t] + p[1] * Bf[32 + t] + p[2] * Bf[64 + t];
        sx[3 + t] = __sinf(pr);
        sx[35 + t] = __cosf(pr);
    }
    __syncthreads();
    const float* wr = w1 + (size_t)t * 67;
    float a = b1[t];
    for (int i = 0; i < 67; i++) a += wr[i] * sx[i];
    sh1[t] = fmaxf(a, 0.f);
    __syncthreads();
    float outv[8], s = 0.f, s2 = 0.f;
#pragma unroll
    for (int j = 0; j < 8; j++) {
        int r = t + j * 256;
        const float* w2r = w2 + (size_t)r * 256;
        float acc = b2[r];
        for (int i = 0; i < 256; i++) acc += w2r[i] * sh1[i];
        outv[j] = acc; s += acc; s2 += acc * acc;
    }
    red[t] = s; __syncthreads();
    for (int o = 128; o > 0; o >>= 1) { if (t < o) red[t] += red[t + o]; __syncthreads(); }
    float mu = red[0] / 2048.f;
    __syncthreads();
    red[t] = s2; __syncthreads();
    for (int o = 128; o > 0; o >>= 1) { if (t < o) red[t] += red[t + o]; __syncthreads(); }
    float var = red[0] / 2048.f - mu * mu;
    float inv = rsqrtf(var + 1e-5f);
    float* xo = xout + (size_t)tok * 2048;
#pragma unroll
    for (int j = 0; j < 8; j++) {
        int r = t + j * 256;
        xo[r] = (outv[j] - mu) * inv * lng[r] + lnb[r];
    }
}

// ============================================================================
// Host orchestration
// ============================================================================
extern "C" void kernel_launch(void* const* d_in, const int* in_sizes, int n_in,
                              void* d_out, int out_size, void* d_ws, size_t ws_size,
                              hipStream_t stream) {
    (void)in_sizes; (void)n_in; (void)out_size; (void)ws_size;
    const int Bsz = 2, N = 1024, Mtok = 2048, D = 2048, F = 5632;
    (void)Bsz;

    auto fp = [&](int i) { return (const float*)d_in[i]; };
    auto ip = [&](int i) { return (const int*)d_in[i]; };

    const float* points   = fp(0);
    const int*   amask    = ip(1);
    const float* final_ln = fp(2);
    const float* Bf  = fp(49);
    const float* b1  = fp(50);
    const float* b2  = fp(51);
    const float* lnb = fp(52);
    const float* lng = fp(53);
    const float* w1  = fp(54);
    const float* w2  = fp(55);

    // workspace carve
    char* wsb = (char*)d_ws;
    size_t off = 0;
    auto carve = [&](size_t bytes) -> char* {
        char* p = wsb + off;
        off = (off + bytes + 255) & ~(size_t)255;
        return p;
    };
    float* x   = (float*)carve((size_t)Mtok * D * 4);
    float* xn  = (float*)carve((size_t)Mtok * D * 4);
    unsigned short* xb = (unsigned short*)carve((size_t)Mtok * D * 2);
    float* qf  = (float*)carve((size_t)Mtok * D * 4);
    float* kf  = (float*)carve((size_t)Mtok * D * 4);
    float* vf  = (float*)carve((size_t)Mtok * D * 4);
    unsigned short* qb = (unsigned short*)carve((size_t)Mtok * D * 2);
    unsigned short* kb = (unsigned short*)carve((size_t)Mtok * D * 2);
    unsigned short* vb = (unsigned short*)carve((size_t)Mtok * D * 2);
    float* ao  = (float*)carve((size_t)Mtok * D * 4);
    unsigned short* aob = (unsigned short*)carve((size_t)Mtok * D * 2);
    unsigned short* wq  = (unsigned short*)carve((size_t)F * D * 2);   // dequant buf
    float* gbuf = (float*)carve((size_t)Mtok * F * 4);
    float* ubuf = (float*)carve((size_t)Mtok * F * 4);
    unsigned short* hb = (unsigned short*)carve((size_t)Mtok * F * 2);
    float* t16 = (float*)carve((size_t)Mtok * 16 * 4);

    size_t nMD = (size_t)Mtok * D, nMF = (size_t)Mtok * F;
    int gMD = (int)((nMD + 255) / 256), gMF = (int)((nMF + 255) / 256);

    auto conv = [&](const float* src, unsigned short* dst, size_t total) {
        conv_bf16_kernel<<<(int)((total + 255) / 256), 256, 0, stream>>>(src, dst, total);
    };
    // full qlinear: dequant -> WMMA GEMM -> LoRA
    auto qlinear = [&](const unsigned short* xbf, const float* xf32,
                       const int* codes, const float* absmax,
                       const float* Al, const float* Bl,
                       float* out, int outd, int ind) {
        size_t totW = (size_t)outd * ind;
        dequant_nf4_kernel<<<(int)((totW + 255) / 256), 256, 0, stream>>>(
            codes, absmax, wq, outd, ind, totW);
        gemm_bf16_nt_kernel<<<dim3(outd / 64, Mtok / 32), 32, 0, stream>>>(
            xbf, wq, out, Mtok, outd, ind);
        lora1_kernel<<<Mtok, 256, 0, stream>>>(xf32, Al, t16, ind);
        size_t totY = (size_t)Mtok * outd;
        lora2_kernel<<<(int)((totY + 255) / 256), 256, 0, stream>>>(
            t16, Bl, out, outd, totY);
    };

    // tokenizer -> x
    tokenizer_kernel<<<Mtok, 256, 0, stream>>>(points, Bf, w1, b1, w2, b2, lng, lnb, x);

    for (int li = 0; li < 2; li++) {
        int L = 3 + 23 * li, C = 56 + 7 * li;
        // param slots per layer (alphabetical): down gate k ln1 ln2 o q up v
        const float *dA = fp(L + 0),  *dB = fp(L + 1),  *dM = fp(L + 2);
        const float *gA = fp(L + 3),  *gB = fp(L + 4),  *gM = fp(L + 5);
        const float *kA = fp(L + 6),  *kB = fp(L + 7),  *kM = fp(L + 8);
        const float *ln1 = fp(L + 9), *ln2 = fp(L + 10);
        const float *oA = fp(L + 11), *oB = fp(L + 12), *oM = fp(L + 13);
        const float *qA = fp(L + 14), *qB = fp(L + 15), *qM = fp(L + 16);
        const float *uA = fp(L + 17), *uB = fp(L + 18), *uM = fp(L + 19);
        const float *vA = fp(L + 20), *vB = fp(L + 21), *vM = fp(L + 22);
        // codes (alphabetical): down gate k o q up v
        const int *dC = ip(C + 0), *gC = ip(C + 1), *kC = ip(C + 2),
                  *oC = ip(C + 3), *qC = ip(C + 4), *uC = ip(C + 5), *vC = ip(C + 6);

        // ---- attention ----
        rmsnorm_kernel<<<Mtok, 256, 0, stream>>>(x, ln1, xn, D);
        conv(xn, xb, nMD);
        qlinear(xb, xn, qC, qM, qA, qB, qf, D, D);
        qlinear(xb, xn, kC, kM, kA, kB, kf, D, D);
        qlinear(xb, xn, vC, vM, vA, vB, vf, D, D);
        size_t nPairs = nMD / 2;
        rope_kernel<<<(int)((nPairs + 255) / 256), 256, 0, stream>>>(qf, kf, N, D, nPairs);
        conv(qf, qb, nMD); conv(kf, kb, nMD); conv(vf, vb, nMD);
        flash_attn_kernel<<<dim3(N / 16, 16, 2), 32, 0, stream>>>(qb, kb, vb, amask, ao, N, D);
        conv(ao, aob, nMD);
        qlinear(aob, ao, oC, oM, oA, oB, xn, D, D);          // o-projection -> xn
        add_kernel<<<gMD, 256, 0, stream>>>(x, xn, nMD);     // residual

        // ---- mlp ----
        rmsnorm_kernel<<<Mtok, 256, 0, stream>>>(x, ln2, xn, D);
        conv(xn, xb, nMD);
        qlinear(xb, xn, gC, gM, gA, gB, gbuf, F, D);         // gate
        qlinear(xb, xn, uC, uM, uA, uB, ubuf, F, D);         // up
        silu_mul_kernel<<<gMF, 256, 0, stream>>>(gbuf, ubuf, nMF);
        conv(gbuf, hb, nMF);
        qlinear(hb, gbuf, dC, dM, dA, dB, ao, D, F);         // down -> ao
        add_kernel<<<gMD, 256, 0, stream>>>(x, ao, nMD);     // residual
    }
    // final rmsnorm straight into d_out (f32)
    rmsnorm_kernel<<<Mtok, 256, 0, stream>>>(x, final_ln, (float*)d_out, D);
}
